// SNNCore_50319836840757
// MI455X (gfx1250) — compile-verified, matrix operand order checked by reading, not run on hardware
//
#include <hip/hip_runtime.h>

#define Bv 8
#define Tv 256
#define Dv 512
#define Sv 256
#define Vv 32000

typedef _Float16 half8 __attribute__((ext_vector_type(8)));
typedef _Float16 v16h  __attribute__((ext_vector_type(16)));
typedef float    v8f   __attribute__((ext_vector_type(8)));

#define WMMA_F16(a, b, c) \
  __builtin_amdgcn_wmma_f32_16x16x32_f16(false, (a), false, (b), (short)0, (c), false, false)

// Build a 16-half WMMA fragment (A or B) from two contiguous 16B chunks.
// Layout per ISA 7.12.2 (f16 16x32 A / 32x16 B, wave32):
//   halfs[0..7]  = K(k0 + hi*8 .. +7),  halfs[8..15] = K(k0+16+hi*8 .. +7)
__device__ __forceinline__ v16h mk_frag(const _Float16* rowbase, int k0, int hi) {
  const half8 lo = *(const half8*)(rowbase + k0 + hi * 8);
  const half8 hj = *(const half8*)(rowbase + k0 + 16 + hi * 8);
  v16h a;
#pragma unroll
  for (int i = 0; i < 8; ++i) { a[i] = lo[i]; a[8 + i] = hj[i]; }
  return a;
}

// ---------------- embedding: x[t][b][d] = tok[ids[b,t]][d] + pos[t][d] ----------------
__global__ void embed_kernel(const int* __restrict__ ids, const float* __restrict__ tok,
                             const float* __restrict__ pos, float* __restrict__ x) {
  for (int i = blockIdx.x * blockDim.x + threadIdx.x; i < Tv * Bv * Dv;
       i += gridDim.x * blockDim.x) {
    int d = i & (Dv - 1);
    int b = (i >> 9) & 7;
    int t = i >> 12;
    x[i] = tok[ids[b * Tv + t] * Dv + d] + pos[t * Dv + d];
  }
}

// ---------------- weight convert + transpose to f16: Wt[n][k] = W[k][n] ----------------
__global__ void convertT_kernel(_Float16* __restrict__ dst, const float* __restrict__ src,
                                int N, int K, int batch, long long srcStride, long long dstStride) {
  long long total = (long long)batch * N * K;
  for (long long i = blockIdx.x * (long long)blockDim.x + threadIdx.x; i < total;
       i += (long long)gridDim.x * blockDim.x) {
    int k = (int)(i % K);
    long long r = i / K;
    int n = (int)(r % N);
    int bb = (int)(r / N);
    dst[bb * dstStride + (long long)n * K + k] = (_Float16)src[bb * srcStride + (long long)k * N + n];
  }
}

// ---------------- the sequential SNN scan: one persistent workgroup, 512 threads ----------------
// 16 waves (4/SIMD -> 256 VGPR budget). enc/gen: 2 N-tiles per wave. inf: 1 N-tile per wave.
// LIF membranes and all per-column parameters are loop-carried registers.
__global__ __launch_bounds__(512)
void scan_kernel(const float* __restrict__ x, float* __restrict__ preds,
                 const _Float16* __restrict__ encWt, const _Float16* __restrict__ genWt,
                 const _Float16* __restrict__ infWt,
                 const float* __restrict__ enc_b, const float* __restrict__ enc_tau,
                 const float* __restrict__ enc_thr,
                 const float* __restrict__ gen_b, const float* __restrict__ gen_tau,
                 const float* __restrict__ gen_thr,
                 const float* __restrict__ inf_b, const float* __restrict__ inf_tau,
                 const float* __restrict__ inf_thr,
                 const float* __restrict__ ne_g, const float* __restrict__ ne_b,
                 const float* __restrict__ ns_g, const float* __restrict__ ns_b) {
  extern __shared__ __align__(16) char smem[];
  _Float16* sA    = (_Float16*)smem;          // [16][512] f16 A staging (rows 8..15 zero)
  float* s_bu     = (float*)(smem + 16384);   // [8][512]
  float* s_pred   = (float*)(smem + 32768);   // [8][512]
  float* s_su     = (float*)(smem + 49152);   // [8][256]
  float* s_states = (float*)(smem + 57344);   // [2][16][256] (rows 8..15 zero)
  float* s_red    = (float*)(smem + 90112);   // [8][2] mean/var

  const int tid = threadIdx.x;
  const int w = tid >> 5, lane = tid & 31, l15 = lane & 15, hi = lane >> 4;

  // ---- persistent register LIF state (fixed (wave,lane) -> column ownership) ----
  float mem_enc_r[2][8] = {};    // [tile u][row]
  float gen_mem_r[2][2][8] = {}; // [j][u][row]
  float inf_mem_r[2][8] = {};    // [j][row]

  const int colD0 = (2 * w) * 16 + l15;
  const int colD1 = (2 * w + 1) * 16 + l15;
  const int colS = w * 16 + l15;  // w<16 always -> < 256

  // ---- hoist per-column parameters (constant over time) into registers ----
  float enc_dec[2], enc_th[2], enc_bi[2];
  float gen_dec[2][2], gen_th[2][2], gen_bi[2][2];
  float inf_dec[2], inf_th[2], inf_bi[2];
  {
    const int cd[2] = {colD0, colD1};
#pragma unroll
    for (int u = 0; u < 2; ++u) {
      enc_dec[u] = __expf(-1.0f / fmaxf(enc_tau[cd[u]], 1.0f));
      enc_th[u] = enc_thr[cd[u]];
      enc_bi[u] = enc_b[cd[u]];
#pragma unroll
      for (int j = 0; j < 2; ++j) {
        gen_dec[j][u] = __expf(-1.0f / fmaxf(gen_tau[j * 512 + cd[u]], 1.0f));
        gen_th[j][u] = gen_thr[j * 512 + cd[u]];
        gen_bi[j][u] = gen_b[j * 512 + cd[u]];
      }
    }
#pragma unroll
    for (int j = 0; j < 2; ++j) {
      inf_dec[j] = __expf(-1.0f / fmaxf(inf_tau[j * 256 + colS], 1.0f));
      inf_th[j] = inf_thr[j * 256 + colS];
      inf_bi[j] = inf_b[j * 256 + colS];
    }
  }
  // ---- LN params: with 512 threads, elementwise loops always hit cc = tid (&mask) ----
  float neg_r[2], neb_r[2], nsg_r[2], nsb_r[2];
#pragma unroll
  for (int j = 0; j < 2; ++j) {
    neg_r[j] = ne_g[j * 512 + tid];         // cc = tid (tid < 512)
    neb_r[j] = ne_b[j * 512 + tid];
    nsg_r[j] = ns_g[j * 256 + (tid & 255)]; // cc = tid & 255
    nsb_r[j] = ns_b[j * 256 + (tid & 255)];
  }

  for (int i = tid; i < 2 * 16 * 256; i += 512) s_states[i] = 0.0f;
  for (int i = tid; i < 16 * 512; i += 512) sA[i] = (_Float16)0.0f;
  __syncthreads();

  for (int t = 0; t < Tv; ++t) {
    // -------- stage x_t (pad M to 16 with zeros), prefetch next timestep --------
    for (int i = tid; i < 16 * 512; i += 512) {
      int r = i >> 9, c = i & 511;
      sA[i] = (r < 8) ? (_Float16)x[t * 4096 + r * 512 + c] : (_Float16)0.0f;
    }
    if (t + 1 < Tv && tid < 128) __builtin_prefetch(x + (t + 1) * 4096 + tid * 32, 0, 1);
    __syncthreads();

    // -------- encoder GEMM (16x512 @ 512x512): 32 N-tiles, 2 per wave --------
    {
      v8f acc[2] = {{}, {}};
      const _Float16* arow = sA + l15 * 512;
      const _Float16* br0 = encWt + (size_t)colD0 * 512;
      const _Float16* br1 = encWt + (size_t)colD1 * 512;
#pragma unroll 4
      for (int k0 = 0; k0 < 512; k0 += 32) {
        v16h a = mk_frag(arow, k0, hi);
        v16h b0 = mk_frag(br0, k0, hi);
        v16h b1 = mk_frag(br1, k0, hi);
        acc[0] = WMMA_F16(a, b0, acc[0]);
        acc[1] = WMMA_F16(a, b1, acc[1]);
      }
      if (hi == 0) {  // lanes 0..15 hold all valid rows M=0..7
        const int cd[2] = {colD0, colD1};
#pragma unroll
        for (int u = 0; u < 2; ++u)
#pragma unroll
          for (int r = 0; r < 8; ++r) {
            float m = mem_enc_r[u][r] * enc_dec[u] + acc[u][r] + enc_bi[u];
            float s = (m >= enc_th[u]) ? 1.0f : 0.0f;
            mem_enc_r[u][r] = m * (1.0f - s);
            s_bu[r * 512 + cd[u]] = s;
          }
      }
    }
    __syncthreads();

    // -------- layers j=0,1 plus extra decode pass (sub==2 -> j=1) --------
#pragma unroll
    for (int sub = 0; sub < 3; ++sub) {
      const int j = (sub == 2) ? 1 : sub;
      const bool extra = (sub == 2);

      // stage states[j] -> sA (K=256); padded rows already zero
      for (int i = tid; i < 16 * 256; i += 512) {
        int r = i >> 8, cc = i & 255;
        sA[r * 512 + cc] = (_Float16)s_states[j * 4096 + i];
      }
      __syncthreads();

      // generative GEMM (16x256 @ 256x512): 32 N-tiles, 2 per wave
      {
        v8f acc[2] = {{}, {}};
        const _Float16* arow = sA + l15 * 512;
        const _Float16* br0 = genWt + j * 512 * 256 + (size_t)colD0 * 256;
        const _Float16* br1 = genWt + j * 512 * 256 + (size_t)colD1 * 256;
#pragma unroll 2
        for (int k0 = 0; k0 < 256; k0 += 32) {
          v16h a = mk_frag(arow, k0, hi);
          v16h b0 = mk_frag(br0, k0, hi);
          v16h b1 = mk_frag(br1, k0, hi);
          acc[0] = WMMA_F16(a, b0, acc[0]);
          acc[1] = WMMA_F16(a, b1, acc[1]);
        }
        if (hi == 0) {
          const int cd[2] = {colD0, colD1};
#pragma unroll
          for (int u = 0; u < 2; ++u)
#pragma unroll
            for (int r = 0; r < 8; ++r) {
              float m = gen_mem_r[j][u][r] * gen_dec[j][u] + acc[u][r] + gen_bi[j][u];
              float s = (m >= gen_th[j][u]) ? 1.0f : 0.0f;
              gen_mem_r[j][u][r] = m * (1.0f - s);
              s_pred[r * 512 + cd[u]] = s;
              if (extra) preds[t * 4096 + r * 512 + cd[u]] = s;  // final_pred output
            }
        }
      }
      __syncthreads();

      // error layernorm stats over D=512 (row r handled by wave r)
      if (w < 8) {
        float sum = 0.f, sq = 0.f;
        for (int cc = lane; cc < 512; cc += 32) {
          float d = extra ? -s_pred[w * 512 + cc] : (s_bu[w * 512 + cc] - s_pred[w * 512 + cc]);
          sum += d; sq += d * d;
        }
        for (int o = 16; o > 0; o >>= 1) { sum += __shfl_xor(sum, o, 32); sq += __shfl_xor(sq, o, 32); }
        if (lane == 0) {
          float mu = sum * (1.0f / 512.0f);
          s_red[w * 2] = mu;
          s_red[w * 2 + 1] = sq * (1.0f / 512.0f) - mu * mu;
        }
      }
      __syncthreads();

      // normalize -> err into sA (f16 A operand) and s_bu (f32, becomes next bu)
#pragma unroll
      for (int k = 0; k < 8; ++k) {
        int i = tid + 512 * k;  // r = k, cc = tid
        float d = extra ? -s_pred[i] : (s_bu[i] - s_pred[i]);
        float e = (d - s_red[k * 2]) * rsqrtf(s_red[k * 2 + 1] + 1e-5f) * neg_r[j] + neb_r[j];
        sA[i] = (_Float16)e;
        if (!extra) s_bu[i] = e;
      }
      __syncthreads();

      // inference GEMM (16x512 @ 512x256): 16 N-tiles, 1 per wave
      {
        v8f acc = {};
        const _Float16* arow = sA + l15 * 512;
        const _Float16* brow = infWt + j * 256 * 512 + (size_t)colS * 512;
#pragma unroll 4
        for (int k0 = 0; k0 < 512; k0 += 32) {
          v16h a = mk_frag(arow, k0, hi);
          v16h b = mk_frag(brow, k0, hi);
          acc = WMMA_F16(a, b, acc);
        }
        if (hi == 0) {
#pragma unroll
          for (int r = 0; r < 8; ++r) {
            float m = inf_mem_r[j][r] * inf_dec[j] + acc[r] + inf_bi[j];
            float s = (m >= inf_th[j]) ? 1.0f : 0.0f;
            inf_mem_r[j][r] = m * (1.0f - s);
            if (!extra) s_su[r * 256 + colS] = s;
          }
        }
      }
      __syncthreads();

      if (!extra) {
        // states layernorm over S=256
        if (w < 8) {
          float sum = 0.f, sq = 0.f;
          for (int cc = lane; cc < 256; cc += 32) {
            float v = s_states[j * 4096 + w * 256 + cc] + s_su[w * 256 + cc];
            sum += v; sq += v * v;
          }
          for (int o = 16; o > 0; o >>= 1) { sum += __shfl_xor(sum, o, 32); sq += __shfl_xor(sq, o, 32); }
          if (lane == 0) {
            float mu = sum * (1.0f / 256.0f);
            s_red[w * 2] = mu;
            s_red[w * 2 + 1] = sq * (1.0f / 256.0f) - mu * mu;
          }
        }
        __syncthreads();
#pragma unroll
        for (int k = 0; k < 4; ++k) {
          int i = tid + 512 * k;  // < 2048
          int r = i >> 8;         // cc = tid & 255
          float v = s_states[j * 4096 + i] + s_su[i];
          s_states[j * 4096 + i] =
              (v - s_red[r * 2]) * rsqrtf(s_red[r * 2 + 1] + 1e-5f) * nsg_r[j] + nsb_r[j];
        }
        __syncthreads();
      }
    }  // sub
  }    // t
}

// ---------------- LayerNorm + exact GeLU -> f16 h, row m = b*T + t ----------------
__global__ void lngelu_kernel(const float* __restrict__ preds, const float* __restrict__ g,
                              const float* __restrict__ b, _Float16* __restrict__ h16) {
  __shared__ float ssum[256], ssq[256];
  const int tid = threadIdx.x;
  const int rid = blockIdx.x;
  const int bb = rid >> 8, t = rid & 255;
  const float* row = preds + t * 4096 + bb * 512;
  float v0 = row[tid], v1 = row[tid + 256];
  ssum[tid] = v0 + v1;
  ssq[tid] = v0 * v0 + v1 * v1;
  __syncthreads();
  for (int o = 128; o > 0; o >>= 1) {
    if (tid < o) { ssum[tid] += ssum[tid + o]; ssq[tid] += ssq[tid + o]; }
    __syncthreads();
  }
  float mu = ssum[0] * (1.0f / 512.0f);
  float var = ssq[0] * (1.0f / 512.0f) - mu * mu;
  float rstd = rsqrtf(var + 1e-5f);
  for (int d = tid; d < 512; d += 256) {
    float xn = (row[d] - mu) * rstd * g[d] + b[d];
    float ge = 0.5f * xn * (1.0f + erff(xn * 0.7071067811865475f));
    h16[(size_t)rid * 512 + d] = (_Float16)ge;
  }
}

// ---------------- big output GEMM: [2048,512]f16 @ [512,32000]f16 -> f32 ----------------
__global__ __launch_bounds__(256)
void outgemm_kernel(const _Float16* __restrict__ h16, const _Float16* __restrict__ outWt,
                    const float* __restrict__ out_b2, float* __restrict__ out) {
  __shared__ _Float16 sA[16 * 512];
  const int tid = threadIdx.x;
  const int w = tid >> 5, lane = tid & 31, l15 = lane & 15, hi = lane >> 4;
  const int m0 = blockIdx.y * 16;
  const int n0 = blockIdx.x * 256 + w * 32;  // each wave: two 16x16 N-tiles

  // Async direct-to-LDS staging of the A tile (CDNA5 ASYNCcnt path: no VGPR round trip).
  for (int i = tid * 8; i < 16 * 512; i += 256 * 8) {
    const _Float16* gsrc = &h16[(size_t)m0 * 512 + i];
    unsigned ldsoff = (unsigned)(unsigned long long)&sA[i];
    asm volatile("global_load_async_to_lds_b128 %0, %1, off"
                 :: "v"(ldsoff), "v"(gsrc)
                 : "memory");
  }
  asm volatile("s_wait_asynccnt 0" ::: "memory");
  __syncthreads();

  v8f c0 = {}, c1 = {};
  const _Float16* arow = sA + l15 * 512;
  const _Float16* b0 = outWt + (size_t)(n0 + l15) * 512;
  const _Float16* b1 = outWt + (size_t)(n0 + 16 + l15) * 512;
#pragma unroll 4
  for (int k0 = 0; k0 < 512; k0 += 32) {
    if (k0 < 256) { __builtin_prefetch(b0 + k0 + 256, 0, 0); __builtin_prefetch(b1 + k0 + 256, 0, 0); }
    v16h a  = mk_frag(arow, k0, hi);
    v16h f0 = mk_frag(b0, k0, hi);
    v16h f1 = mk_frag(b1, k0, hi);
    c0 = WMMA_F16(a, f0, c0);
    c1 = WMMA_F16(a, f1, c1);
  }
  const int M0 = m0 + hi * 8;
  const float bb0 = out_b2[n0 + l15], bb1 = out_b2[n0 + 16 + l15];
#pragma unroll
  for (int r = 0; r < 8; ++r) {
    size_t rowo = (size_t)(M0 + r) * Vv;
    out[rowo + n0 + l15] = c0[r] + bb0;
    out[rowo + n0 + 16 + l15] = c1[r] + bb1;
  }
}

extern "C" void kernel_launch(void* const* d_in, const int* in_sizes, int n_in,
                              void* d_out, int out_size, void* d_ws, size_t ws_size,
                              hipStream_t stream) {
  const int*   ids     = (const int*)  d_in[0];
  const float* tok_emb = (const float*)d_in[1];
  const float* pos_emb = (const float*)d_in[2];
  const float* enc_W   = (const float*)d_in[3];
  const float* enc_b   = (const float*)d_in[4];
  const float* enc_tau = (const float*)d_in[5];
  const float* enc_thr = (const float*)d_in[6];
  const float* gen_W   = (const float*)d_in[7];
  const float* gen_b   = (const float*)d_in[8];
  const float* gen_tau = (const float*)d_in[9];
  const float* gen_thr = (const float*)d_in[10];
  const float* inf_W   = (const float*)d_in[11];
  const float* inf_b   = (const float*)d_in[12];
  const float* inf_tau = (const float*)d_in[13];
  const float* inf_thr = (const float*)d_in[14];
  const float* ne_g    = (const float*)d_in[15];
  const float* ne_b    = (const float*)d_in[16];
  const float* ns_g    = (const float*)d_in[17];
  const float* ns_b    = (const float*)d_in[18];
  const float* out_g   = (const float*)d_in[19];
  const float* out_b   = (const float*)d_in[20];
  const float* out_W   = (const float*)d_in[21];
  const float* out_b2  = (const float*)d_in[22];

  char* ws = (char*)d_ws;
  float*    x_ws  = (float*)(ws);                        // [T][8][512] f32, 4MB
  float*    preds = (float*)(ws + (4 << 20));            // [T][8][512] f32, 4MB
  _Float16* h16   = (_Float16*)(ws + (8 << 20));         // [2048][512] f16, 2MB
  _Float16* encWt = (_Float16*)(ws + (8 << 20) + (2 << 20));
  _Float16* genWt = encWt + 512 * 512;
  _Float16* infWt = genWt + 2 * 512 * 256;
  _Float16* outWt = infWt + 2 * 256 * 512;               // [32000][512] f16, 32MB

  embed_kernel<<<4096, 256, 0, stream>>>(ids, tok_emb, pos_emb, x_ws);
  convertT_kernel<<<1024, 256, 0, stream>>>(encWt, enc_W, 512, 512, 1, 0, 0);
  convertT_kernel<<<1024, 256, 0, stream>>>(genWt, gen_W, 512, 256, 2, 256LL * 512, 512LL * 256);
  convertT_kernel<<<1024, 256, 0, stream>>>(infWt, inf_W, 256, 512, 2, 512LL * 256, 256LL * 512);
  convertT_kernel<<<8192, 256, 0, stream>>>(outWt, out_W, 32000, 512, 1, 0, 0);
  scan_kernel<<<1, 512, 90176, stream>>>(x_ws, preds, encWt, genWt, infWt,
                                         enc_b, enc_tau, enc_thr, gen_b, gen_tau, gen_thr,
                                         inf_b, inf_tau, inf_thr, ne_g, ne_b, ns_g, ns_b);
  lngelu_kernel<<<2048, 256, 0, stream>>>(preds, out_g, out_b, h16);
  outgemm_kernel<<<dim3(125, 128), 256, 0, stream>>>(h16, outWt, out_b2, (float*)d_out);
}
